// CausalSelfAttention_12025908428910
// MI455X (gfx1250) — matrix operand order repaired and checked by reading, not compiled, not run on hardware
//
#include <hip/hip_runtime.h>
#include <hip/hip_bf16.h>

// ---------------------------------------------------------------------------
// CDNA5 (gfx1250) attention block:  out = proj( attn( x @ w_attn ) @ w_proj )
// B=4, N=2048, C=1024, H=16, D=64.  All WMMA in bf16 with f32 accumulate.
// ---------------------------------------------------------------------------

typedef __attribute__((ext_vector_type(16))) __bf16 v16bf;
typedef __attribute__((ext_vector_type(8)))  float  v8f;
typedef __attribute__((ext_vector_type(4)))  int    v4i;

#define B_   4
#define N_   2048
#define C_   1024
#define H_   16
#define D_   64
#define MN_  (B_ * N_)        // 8192 rows
#define LDQKV_ (3 * C_)       // 3072
#define LOG2E_ 1.4426950408889634f
#define SCALE_ 0.125f         // 1/sqrt(64)

#if defined(__AMDGCN__) && __has_builtin(__builtin_amdgcn_global_load_async_to_lds_b128)
#define HAVE_ASYNC_LDS 1
#else
#define HAVE_ASYNC_LDS 0
#endif

// raw v_exp_f32 (no libm denormal guard; softmax wants underflow -> 0)
#if defined(__AMDGCN__) && __has_builtin(__builtin_amdgcn_exp2f)
#define EXP2F(x) __builtin_amdgcn_exp2f(x)
#else
#define EXP2F(x) exp2f(x)
#endif

// ---------------- WMMA helpers ----------------

__device__ inline v8f wmma_bf16(v16bf a, v16bf b, v8f c) {
    return __builtin_amdgcn_wmma_f32_16x16x32_bf16(
        /*neg_a=*/false, a, /*neg_b=*/false, b,
        /*c_mod=*/(short)0, c, /*reuse_a=*/false, /*reuse_b=*/false);
}

// A-matrix fragment (16x32 bf16, row-major source [row][k], row stride ld).
// Lane half h picks K = {h*8..h*8+7, 16+h*8..16+h*8+7}  -> two 16B loads.
__device__ inline v16bf load_frag_a(const __bf16* base, int ld, int lane) {
    int row = lane & 15;
    int h   = lane >> 4;
    const __bf16* p = base + row * ld + h * 8;
    union { uint4 q[2]; v16bf v; } u;
    u.q[0] = *(const uint4*)(p);
    u.q[1] = *(const uint4*)(p + 16);
    return u.v;
}

// B-matrix fragment (32x16 bf16) from K-transposed storage BT[n][k], stride ld.
// Lane half h picks K = {h*8..h*8+7, 16+h*8..16+h*8+7} for column n = lane&15 -> two 16B loads.
__device__ inline v16bf load_frag_b(const __bf16* base, int ld, int lane) {
    int n = lane & 15;
    int h = lane >> 4;
    const __bf16* p = base + n * ld + h * 8;
    union { uint4 q[2]; v16bf v; } u;
    u.q[0] = *(const uint4*)(p);
    u.q[1] = *(const uint4*)(p + 16);
    return u.v;
}

// ---------------- DPP 16-lane butterfly reductions (stay inside lane half) --

template <int CTRL>
__device__ inline float dpp_f32(float x) {
    int r = __builtin_amdgcn_update_dpp(
        0, __builtin_bit_cast(int, x), CTRL, 0xf, 0xf, true);
    return __builtin_bit_cast(float, r);
}
// xor1, xor2, half-mirror (i^7), mirror (i^15): reduces each 16-lane row.
__device__ inline float rowmax16(float v) {
    v = fmaxf(v, dpp_f32<0xB1>(v));    // quad_perm(1,0,3,2)
    v = fmaxf(v, dpp_f32<0x4E>(v));    // quad_perm(2,3,0,1)
    v = fmaxf(v, dpp_f32<0x141>(v));   // row_half_mirror
    v = fmaxf(v, dpp_f32<0x140>(v));   // row_mirror
    return v;
}
__device__ inline float rowsum16(float v) {
    v += dpp_f32<0xB1>(v);
    v += dpp_f32<0x4E>(v);
    v += dpp_f32<0x141>(v);
    v += dpp_f32<0x140>(v);
    return v;
}

// ---------------- async / staging helpers ----------------

__device__ inline void copy16_g2l(const __bf16* g, __bf16* l) {
#if HAVE_ASYNC_LDS
    __builtin_amdgcn_global_load_async_to_lds_b128(
        (__attribute__((address_space(1))) v4i*)g,
        (__attribute__((address_space(3))) v4i*)l, 0, 0);
#else
    *(uint4*)l = *(const uint4*)g;
#endif
}
__device__ inline void wait_async_done() {
#if HAVE_ASYNC_LDS
    asm volatile("s_wait_asynccnt 0x0" ::: "memory");
#endif
}

struct bf16x4 { union { __bf16 h[4]; uint2 u; }; };

__device__ inline bf16x4 cvt4(const float* p) {
    float4 f = *(const float4*)p;
    bf16x4 r;
    r.h[0] = (__bf16)f.x; r.h[1] = (__bf16)f.y;
    r.h[2] = (__bf16)f.z; r.h[3] = (__bf16)f.w;
    return r;
}
__device__ inline bf16x4 cvt4(const __bf16* p) {
    bf16x4 r;
    r.u = *(const uint2*)p;
    return r;
}

__device__ inline void store_out(float*  p, float v) { *p = v; }
__device__ inline void store_out(__bf16* p, float v) { *p = (__bf16)v; }

// ---------------------------------------------------------------------------
// Tiled GEMM:  C[M,N] = A[M,K] * B[K,N].  A is f32 or bf16, B is f32,
// C is f32 or bf16.  Tile: BM=128, BN=128, BK=64.  256 threads = 8 waves,
// wave grid 4x2, each wave owns a 32x64 sub-tile (2x4 WMMA tiles).
// Global loads for tile t+1 are issued into registers while tile t computes.
// M % 128 == 0, N % 128 == 0, K % 64 == 0 (true for all uses here).
// ---------------------------------------------------------------------------
template <typename TA, typename TOut>
__global__ __launch_bounds__(256)
void gemm_wmma(const TA* __restrict__ A, const float* __restrict__ Bm,
               TOut* __restrict__ Cm, int M, int N, int K) {
    __shared__ alignas(16) __bf16 Asl[128][64];   // [m][k]
    __shared__ alignas(16) __bf16 Btl[128][64];   // [n][k]  (transposed)

    const int tid  = threadIdx.x;
    const int lane = tid & 31;
    const int wave = tid >> 5;
    const int wm   = wave >> 1;          // 0..3  -> 32 rows each
    const int wn   = wave & 1;           // 0..1  -> 64 cols each
    const int blockM = blockIdx.y * 128;
    const int blockN = blockIdx.x * 128;

    v8f acc[2][4] = {};
    bf16x4 aBuf[8], bBuf[8];             // register-staged next tile

    // issue global loads for one K-step: 8 chunks of 4 elements each for A & B
    auto gload = [&](int k0) {
        for (int j = 0; j < 8; ++j) {
            int idx = (tid + j * 256) * 4;
            int r = idx >> 6, c = idx & 63;          // A: [128][64]
            aBuf[j] = cvt4(A + (size_t)(blockM + r) * K + k0 + c);
            int n = idx & 127, kk = idx >> 7;        // B: [64][128]
            float4 f = *(const float4*)(Bm + (size_t)(k0 + kk) * N + blockN + n);
            bf16x4 t;
            t.h[0] = (__bf16)f.x; t.h[1] = (__bf16)f.y;
            t.h[2] = (__bf16)f.z; t.h[3] = (__bf16)f.w;
            bBuf[j] = t;
        }
    };
    // commit register-staged tile to LDS
    auto lstore = [&]() {
        for (int j = 0; j < 8; ++j) {
            int idx = (tid + j * 256) * 4;
            int r = idx >> 6, c = idx & 63;
            *(uint2*)&Asl[r][c] = aBuf[j].u;
            int n = idx & 127, kk = idx >> 7;
            Btl[n + 0][kk] = bBuf[j].h[0];
            Btl[n + 1][kk] = bBuf[j].h[1];
            Btl[n + 2][kk] = bBuf[j].h[2];
            Btl[n + 3][kk] = bBuf[j].h[3];
        }
    };

    gload(0);
    for (int k0 = 0; k0 < K; k0 += 64) {
        lstore();
        __syncthreads();
        if (k0 + 64 < K) gload(k0 + 64);   // overlap next loads with compute

        for (int kc = 0; kc < 64; kc += 32) {
            v16bf af[2], bf4[4];
            for (int mt = 0; mt < 2; ++mt)
                af[mt] = load_frag_a(&Asl[wm * 32 + mt * 16][kc], 64, lane);
            for (int nt = 0; nt < 4; ++nt)
                bf4[nt] = load_frag_b(&Btl[wn * 64 + nt * 16][kc], 64, lane);
            for (int mt = 0; mt < 2; ++mt)
                for (int nt = 0; nt < 4; ++nt)
                    acc[mt][nt] = wmma_bf16(af[mt], bf4[nt], acc[mt][nt]);
        }
        __syncthreads();                   // LDS reusable for next lstore
    }

    // epilogue: C/D layout row = r + (lane>>4)*8, col = lane&15
    const int half = lane >> 4;
    const int col0 = lane & 15;
    for (int mt = 0; mt < 2; ++mt)
        for (int nt = 0; nt < 4; ++nt)
            for (int r = 0; r < 8; ++r) {
                int row = blockM + wm * 32 + mt * 16 + half * 8 + r;
                int col = blockN + wn * 64 + nt * 16 + col0;
                store_out(&Cm[(size_t)row * N + col], acc[mt][nt][r]);
            }
}

// ---------------------------------------------------------------------------
// Flash attention, non-causal.  qkv: bf16 [B*N][3C], y: bf16 [B*N][C].
// Grid: (N/64, B*H).  Block: 128 threads = 4 waves, each wave owns 16
// query rows.  Double-buffered K/V tiles: async K loads for tile t+1 are
// issued before computing tile t, hidden behind 16 WMMAs + softmax.
// SCALE*LOG2E is folded into Q so scores come out in log2 domain:
// p = exp2(s - m), alpha = exp2(m_old - m_new), raw v_exp_f32.
// ---------------------------------------------------------------------------
__global__ __launch_bounds__(128)
void attn_flash(const __bf16* __restrict__ qkv, __bf16* __restrict__ y) {
    __shared__ alignas(16) __bf16 Kl[2][64][64];     // [buf][key][d]
    __shared__ alignas(16) __bf16 VTl[2][64][64];    // [buf][d][key]
    __shared__ alignas(16) __bf16 Pl[4][16][64];     // per-wave P tile

    const int tid  = threadIdx.x;
    const int lane = tid & 31;
    const int wave = tid >> 5;                        // 0..3
    const int b    = blockIdx.y / H_;
    const int h    = blockIdx.y % H_;
    const int qTile = blockIdx.x;                     // 0..31

    const int qRow0 = b * N_ + qTile * 64 + wave * 16;  // this wave's 16 rows
    const int qCol0 = h * D_;

    // Q fragments persist across all key tiles; fold SCALE*LOG2E in once so
    // the softmax works directly in the log2 domain (no per-element muls).
    v16bf aq[2];
    aq[0] = load_frag_a(qkv + (size_t)qRow0 * LDQKV_ + qCol0 + 0,  LDQKV_, lane);
    aq[1] = load_frag_a(qkv + (size_t)qRow0 * LDQKV_ + qCol0 + 32, LDQKV_, lane);
    for (int kc = 0; kc < 2; ++kc)
        for (int e = 0; e < 16; ++e)
            aq[kc][e] = (__bf16)((float)aq[kc][e] * (SCALE_ * LOG2E_));

    const int half = lane >> 4;

    // stage key/value tile kt into buffer buf (K async, V transposed manual)
    auto stage_tile = [&](int kt, int buf) {
        const int kRow0 = b * N_ + kt * 64;
        for (int i = tid; i < (64 * 64) / 8; i += 128) {
            int idx = i * 8;
            int r = idx >> 6, c = idx & 63;
            copy16_g2l(qkv + (size_t)(kRow0 + r) * LDQKV_ + C_ + qCol0 + c,
                       &Kl[buf][r][c]);
        }
        for (int i = tid; i < (64 * 64) / 8; i += 128) {
            int idx = i * 8;
            int r = idx >> 6, c = idx & 63;
            union { uint4 q; __bf16 hh[8]; } u;
            u.q = *(const uint4*)(qkv + (size_t)(kRow0 + r) * LDQKV_ + 2 * C_ + qCol0 + c);
            for (int j = 0; j < 8; ++j) VTl[buf][c + j][r] = u.hh[j];
        }
    };

    float m[8], l[8];
    for (int r = 0; r < 8; ++r) { m[r] = -__builtin_inff(); l[r] = 0.f; }
    v8f o[4] = {};

    stage_tile(0, 0);
    wait_async_done();
    __syncthreads();

    for (int kt = 0; kt < N_ / 64; ++kt) {
        const int cur = kt & 1;
        if (kt + 1 < N_ / 64) stage_tile(kt + 1, cur ^ 1);  // prefetch next

        // S = Q * K^T  (per wave: 16 x 64 log2-domain scores, 4 accum tiles)
        v8f s[4] = {};
        for (int nt = 0; nt < 4; ++nt)
            for (int kc = 0; kc < 2; ++kc) {
                v16bf bf = load_frag_b(&Kl[cur][nt * 16][kc * 32], 64, lane);
                s[nt] = wmma_bf16(aq[kc], bf, s[nt]);
            }

        // online softmax update (DPP butterflies, raw v_exp_f32)
        float alpha[8];
        for (int r = 0; r < 8; ++r) {
            float v = fmaxf(fmaxf(s[0][r], s[1][r]), fmaxf(s[2][r], s[3][r]));
            v = rowmax16(v);
            float mnew = fmaxf(m[r], v);
            alpha[r] = EXP2F(m[r] - mnew);
            m[r] = mnew;
        }
        for (int r = 0; r < 8; ++r) {
            float acc_p = 0.f;
            for (int nt = 0; nt < 4; ++nt) {
                float p = EXP2F(s[nt][r] - m[r]);
                s[nt][r] = p;
                acc_p += p;
            }
            l[r] = l[r] * alpha[r] + rowsum16(acc_p);
        }
        for (int dt = 0; dt < 4; ++dt)
            for (int r = 0; r < 8; ++r) o[dt][r] *= alpha[r];

        // re-layout P: C-fragment -> LDS row-major [16][64], then read as A
        for (int nt = 0; nt < 4; ++nt)
            for (int r = 0; r < 8; ++r)
                Pl[wave][half * 8 + r][nt * 16 + (lane & 15)] = (__bf16)s[nt][r];
        // same-wave LDS ops are in-order; fence the compiler + wait DScnt
        asm volatile("s_wait_dscnt 0" ::: "memory");

        // O += P * V
        for (int kc = 0; kc < 2; ++kc) {
            v16bf ap = load_frag_a(&Pl[wave][0][kc * 32], 64, lane);
            for (int dt = 0; dt < 4; ++dt) {
                v16bf bf = load_frag_b(&VTl[cur][dt * 16][kc * 32], 64, lane);
                o[dt] = wmma_bf16(ap, bf, o[dt]);
            }
        }

        wait_async_done();    // next tile's async loads landed
        __syncthreads();      // + all waves done with buffers for this kt
    }

    // normalize and store y[b,n, h*64 + d]  (bf16)
    for (int r = 0; r < 8; ++r) {
        float inv_l = 1.f / l[r];
        int row = qRow0 + half * 8 + r;
        for (int dt = 0; dt < 4; ++dt)
            y[(size_t)row * C_ + qCol0 + dt * 16 + (lane & 15)] =
                (__bf16)(o[dt][r] * inv_l);
    }
}

// ---------------------------------------------------------------------------

extern "C" void kernel_launch(void* const* d_in, const int* in_sizes, int n_in,
                              void* d_out, int out_size, void* d_ws, size_t ws_size,
                              hipStream_t stream) {
    (void)in_sizes; (void)n_in; (void)out_size; (void)ws_size;
    const float* x      = (const float*)d_in[0];   // [B,N,C]
    const float* w_attn = (const float*)d_in[1];   // [C,3C]
    const float* w_proj = (const float*)d_in[2];   // [C,C]
    float* out = (float*)d_out;                    // [B,N,C]

    __bf16* qkv = (__bf16*)d_ws;                       // [MN][3C] bf16 (48 MB)
    __bf16* yb  = qkv + (size_t)MN_ * LDQKV_;          // [MN][C]  bf16 (16 MB)

    // 1) qkv = x @ w_attn   (f32 x f32 -> bf16)
    gemm_wmma<float, __bf16><<<dim3(LDQKV_ / 128, MN_ / 128), 256, 0, stream>>>(
        x, w_attn, qkv, MN_, LDQKV_, C_);

    // 2) y = softmax(q k^T / sqrt(D)) v   (bf16 -> bf16)
    attn_flash<<<dim3(N_ / 64, B_ * H_), 128, 0, stream>>>(qkv, yb);

    // 3) out = y @ w_proj   (bf16 x f32 -> f32)
    gemm_wmma<__bf16, float><<<dim3(C_ / 128, MN_ / 128), 256, 0, stream>>>(
        yb, w_proj, out, MN_, C_, C_);
}
